// MaskedMultiHeadAttention_35132832481476
// MI455X (gfx1250) — matrix operand order repaired and checked by reading, not compiled, not run on hardware
//
#include <hip/hip_runtime.h>
#include <hip/hip_bf16.h>

// ---------------------------------------------------------------------------
// Types for CDNA5 WMMA (wave32): A/B = 16 bf16 per lane, C/D = 8 f32 per lane.
// ---------------------------------------------------------------------------
typedef __bf16        v16bf __attribute__((ext_vector_type(16)));
typedef float         v8f   __attribute__((ext_vector_type(8)));
typedef unsigned int  v4u   __attribute__((ext_vector_type(4)));

union FragAB {
    v16bf bf;
    v4u   q[2];
};

__device__ __forceinline__ unsigned short f32_to_bf16bits(float f) {
    unsigned u = __float_as_uint(f);
    return (unsigned short)((u + 0x7FFFu + ((u >> 16) & 1u)) >> 16);  // RNE
}

// A-fragment, 16x32 (MxK), source row-major with leading dim `ld` (elements).
// ISA layout: lane m = lane&15, half = lane>>4; V0..3 hold K = 8*half..+7,
// V4..7 hold K = 16+8*half..+7  -> two 16B loads.
__device__ __forceinline__ FragAB load_frag_A(const unsigned short* base, int ld,
                                              int m0, int k0) {
    FragAB f;
    int lane = threadIdx.x & 31;
    const unsigned short* p =
        base + (size_t)(m0 + (lane & 15)) * ld + k0 + 8 * (lane >> 4);
    f.q[0] = *(const v4u*)(p);
    f.q[1] = *(const v4u*)(p + 16);
    return f;
}

// B-fragment, 32x16 (KxN), source n-major: element [n][k] at n*ld + k.
// ISA layout: lane n = lane&15, half = lane>>4; V0..3 hold K = 8*half..+7,
// V4..7 hold K = 16+8*half..+7  -> two 16B loads.
__device__ __forceinline__ FragAB load_frag_B(const unsigned short* base, int ld,
                                              int n0, int k0) {
    FragAB f;
    int lane = threadIdx.x & 31;
    const unsigned short* p =
        base + (size_t)(n0 + (lane & 15)) * ld + k0 + 8 * (lane >> 4);
    f.q[0] = *(const v4u*)(p);
    f.q[1] = *(const v4u*)(p + 16);
    return f;
}

__device__ __forceinline__ v8f wmma_bf16(const FragAB& a, const FragAB& b, v8f c) {
    return __builtin_amdgcn_wmma_f32_16x16x32_bf16(
        /*neg_a=*/false, a.bf, /*neg_b=*/false, b.bf,
        /*c_mod=*/(short)0, c, /*reuse_a=*/false, /*reuse_b=*/false);
}

__device__ __forceinline__ v8f zero8() {
    v8f z;
#pragma unroll
    for (int i = 0; i < 8; ++i) z[i] = 0.0f;
    return z;
}

// Row reductions over the 16 lanes of a half-wave (xor masks 1..8 never cross
// the lane-15/16 boundary, matching the C/D layout's row replication).
__device__ __forceinline__ float rowmax16(float x) {
#pragma unroll
    for (int m = 8; m >= 1; m >>= 1) x = fmaxf(x, __shfl_xor(x, m, 32));
    return x;
}
__device__ __forceinline__ float rowsum16(float x) {
#pragma unroll
    for (int m = 8; m >= 1; m >>= 1) x += __shfl_xor(x, m, 32);
    return x;
}

#define D_MODEL 1024
#define SEQ     2048
#define NH      16
#define HD      64
#define BATCH   2

// ---------------------------------------------------------------------------
// 1) f32 -> bf16 elementwise pack
// ---------------------------------------------------------------------------
__global__ void pack_bf16_kernel(const float* __restrict__ in,
                                 unsigned short* __restrict__ out, int n) {
    int i = blockIdx.x * blockDim.x + threadIdx.x;
    if (i < n) out[i] = f32_to_bf16bits(in[i]);
}

// ---------------------------------------------------------------------------
// 2) batched transpose + pack: in[b][R][C] f32 -> out[b][C][R] bf16
//    block = 256 (32x8), 32x32 LDS tile
// ---------------------------------------------------------------------------
__global__ void transpose_pack_kernel(const float* __restrict__ in,
                                      unsigned short* __restrict__ out,
                                      int R, int C) {
    __shared__ float tile[32][33];
    int batch = blockIdx.z;
    int r0 = blockIdx.y * 32, c0 = blockIdx.x * 32;
    in  += (size_t)batch * R * C;
    out += (size_t)batch * R * C;
    int tx = threadIdx.x & 31;
    int ty = threadIdx.x >> 5;
#pragma unroll
    for (int i = 0; i < 32; i += 8) {
        int r = r0 + ty + i, c = c0 + tx;
        tile[ty + i][tx] = (r < R && c < C) ? in[(size_t)r * C + c] : 0.0f;
    }
    __syncthreads();
#pragma unroll
    for (int i = 0; i < 32; i += 8) {
        int c = c0 + ty + i, r = r0 + tx;
        if (c < C && r < R) out[(size_t)c * R + r] = f32_to_bf16bits(tile[tx][ty + i]);
    }
}

// bf16 variant (for V -> V^T)
__global__ void transpose_bf16_kernel(const unsigned short* __restrict__ in,
                                      unsigned short* __restrict__ out,
                                      int R, int C) {
    __shared__ unsigned short tile[32][33];
    int batch = blockIdx.z;
    int r0 = blockIdx.y * 32, c0 = blockIdx.x * 32;
    in  += (size_t)batch * R * C;
    out += (size_t)batch * R * C;
    int tx = threadIdx.x & 31;
    int ty = threadIdx.x >> 5;
#pragma unroll
    for (int i = 0; i < 32; i += 8) {
        int r = r0 + ty + i, c = c0 + tx;
        tile[ty + i][tx] = (r < R && c < C) ? in[(size_t)r * C + c] : (unsigned short)0;
    }
    __syncthreads();
#pragma unroll
    for (int i = 0; i < 32; i += 8) {
        int c = c0 + ty + i, r = r0 + tx;
        if (c < C && r < R) out[(size_t)c * R + r] = tile[tx][ty + i];
    }
}

// ---------------------------------------------------------------------------
// 3) QKV projection: per wave one (proj, b, h, mtile) -> 16x64 tile.
//    Software-pipelined: next K-chunk's fragments are loaded into distinct
//    registers while the current chunk's 4 WMMAs issue, so the compiler can
//    emit partial s_wait_loadcnt instead of a full drain before each WMMA.
// ---------------------------------------------------------------------------
__global__ void __launch_bounds__(256)
qkv_kernel(const unsigned short* __restrict__ Xbf,
           const unsigned short* __restrict__ WqT,
           const unsigned short* __restrict__ WkT,
           const unsigned short* __restrict__ WvT,
           const float* __restrict__ bq, const float* __restrict__ bk,
           const float* __restrict__ bv,
           unsigned short* __restrict__ Q, unsigned short* __restrict__ K,
           unsigned short* __restrict__ V) {
    int wave = blockIdx.x * (blockDim.x >> 5) + (threadIdx.x >> 5);
    int mt = wave & 127;           // S/16 tiles
    int h  = (wave >> 7) & 15;
    int b  = (wave >> 11) & 1;
    int p  = wave >> 12;           // 0=q, 1=k, 2=v

    const unsigned short* WT   = (p == 0) ? WqT : (p == 1) ? WkT : WvT;
    const float*          bias = (p == 0) ? bq  : (p == 1) ? bk  : bv;
    unsigned short*       out  = (p == 0) ? Q   : (p == 1) ? K   : V;

    const unsigned short* x  = Xbf + (size_t)b * SEQ * D_MODEL;
    const unsigned short* wt = WT  + (size_t)h * HD * D_MODEL;
    int m0 = mt * 16;

    v8f acc[4];
#pragma unroll
    for (int nt = 0; nt < 4; ++nt) acc[nt] = zero8();

    FragAB a  = load_frag_A(x, D_MODEL, m0, 0);
    FragAB b0 = load_frag_B(wt, D_MODEL, 0,  0);
    FragAB b1 = load_frag_B(wt, D_MODEL, 16, 0);
    FragAB b2 = load_frag_B(wt, D_MODEL, 32, 0);
    FragAB b3 = load_frag_B(wt, D_MODEL, 48, 0);

    for (int k0 = 0; k0 < D_MODEL; k0 += 32) {
        int kn = (k0 + 32 < D_MODEL) ? k0 + 32 : k0;  // stay in bounds
        FragAB an = load_frag_A(x, D_MODEL, m0, kn);
        FragAB c0 = load_frag_B(wt, D_MODEL, 0,  kn);
        FragAB c1 = load_frag_B(wt, D_MODEL, 16, kn);
        FragAB c2 = load_frag_B(wt, D_MODEL, 32, kn);
        FragAB c3 = load_frag_B(wt, D_MODEL, 48, kn);
        acc[0] = wmma_bf16(a, b0, acc[0]);
        acc[1] = wmma_bf16(a, b1, acc[1]);
        acc[2] = wmma_bf16(a, b2, acc[2]);
        acc[3] = wmma_bf16(a, b3, acc[3]);
        a = an; b0 = c0; b1 = c1; b2 = c2; b3 = c3;
    }

    int lane = threadIdx.x & 31;
    int nn = lane & 15, half = lane >> 4;
#pragma unroll
    for (int nt = 0; nt < 4; ++nt) {
        float bval = bias[h * HD + nt * 16 + nn];
#pragma unroll
        for (int r = 0; r < 8; ++r) {
            int row = m0 + r + 8 * half;
            out[((size_t)(b * NH + h) * SEQ + row) * HD + nt * 16 + nn] =
                f32_to_bf16bits(acc[nt][r] + bval);
        }
    }
}

// ---------------------------------------------------------------------------
// 4) Fused causal flash attention. One wave = one (b,h, 16-row q-tile).
//    Streams 32-key chunks: 4 WMMAs for S = q k^T, online softmax in the C/D
//    layout, P re-laid-out via wave-private LDS, 4 WMMAs for O += P V.
//    Pipelined: current chunk's V fragments and next chunk's K fragments are
//    issued before the softmax VALU sequence, which then covers their latency.
// ---------------------------------------------------------------------------
__global__ void __launch_bounds__(256)
attn_kernel(const unsigned short* __restrict__ Q,
            const unsigned short* __restrict__ K,
            const unsigned short* __restrict__ Vt,
            unsigned short* __restrict__ O) {
    __shared__ __align__(16) unsigned short plds[8 * 16 * 32];  // 1KB per wave
    int widx = threadIdx.x >> 5;
    int wave = blockIdx.x * (blockDim.x >> 5) + widx;
    int mt = wave & 127;
    int bh = wave >> 7;            // 0..31
    int h  = bh & 15;
    int b  = bh >> 4;

    const unsigned short* q  = Q  + (size_t)bh * SEQ * HD;
    const unsigned short* k  = K  + (size_t)bh * SEQ * HD;
    const unsigned short* vt = Vt + (size_t)bh * HD * SEQ;
    unsigned short* lp = plds + widx * (16 * 32);

    int m0 = mt * 16;
    int lane = threadIdx.x & 31;
    int nn = lane & 15, half = lane >> 4;
    const float scl = 0.125f;                  // 1/sqrt(64)
    const float NEG_INF = -__builtin_inff();

    FragAB qa0 = load_frag_A(q, HD, m0, 0);    // e = 0..31
    FragAB qa1 = load_frag_A(q, HD, m0, 32);   // e = 32..63

    v8f acc[4];
#pragma unroll
    for (int nt = 0; nt < 4; ++nt) acc[nt] = zero8();
    float mstate[8], lstate[8];
#pragma unroll
    for (int r = 0; r < 8; ++r) { mstate[r] = NEG_INF; lstate[r] = 0.0f; }

    // prologue: K fragments for the first chunk
    FragAB kb0 = load_frag_B(k, HD, 0,  0);
    FragAB kb1 = load_frag_B(k, HD, 0,  32);
    FragAB kb2 = load_frag_B(k, HD, 16, 0);
    FragAB kb3 = load_frag_B(k, HD, 16, 32);

    for (int jc = 0; jc < m0 + 16; jc += 32) {
        v8f s0 = zero8(), s1 = zero8();
        s0 = wmma_bf16(qa0, kb0, s0);
        s0 = wmma_bf16(qa1, kb1, s0);
        s1 = wmma_bf16(qa0, kb2, s1);
        s1 = wmma_bf16(qa1, kb3, s1);

        // V fragments for this chunk (consumed after the softmax)
        FragAB vb0 = load_frag_B(vt, SEQ, 0,  jc);
        FragAB vb1 = load_frag_B(vt, SEQ, 16, jc);
        FragAB vb2 = load_frag_B(vt, SEQ, 32, jc);
        FragAB vb3 = load_frag_B(vt, SEQ, 48, jc);

        // K fragments for the next chunk
        int jn = (jc + 32 < m0 + 16) ? jc + 32 : jc;
        kb0 = load_frag_B(k, HD, jn,      0);
        kb1 = load_frag_B(k, HD, jn,      32);
        kb2 = load_frag_B(k, HD, jn + 16, 0);
        kb3 = load_frag_B(k, HD, jn + 16, 32);

        bool needmask = (jc + 31 > m0);        // wave-uniform
#pragma unroll
        for (int r = 0; r < 8; ++r) {
            int row = m0 + r + 8 * half;
            float a0 = s0[r] * scl;
            float a1 = s1[r] * scl;
            if (needmask) {
                a0 = (jc + nn      > row) ? NEG_INF : a0;
                a1 = (jc + 16 + nn > row) ? NEG_INF : a1;
            }
            float mx    = rowmax16(fmaxf(a0, a1));
            float mnew  = fmaxf(mstate[r], mx);
            float alpha = __expf(mstate[r] - mnew);
            float p0    = __expf(a0 - mnew);
            float p1    = __expf(a1 - mnew);
            lstate[r] = lstate[r] * alpha + rowsum16(p0 + p1);
            mstate[r] = mnew;
#pragma unroll
            for (int nt = 0; nt < 4; ++nt) acc[nt][r] = acc[nt][r] * alpha;
            lp[(r + 8 * half) * 32 + nn]      = f32_to_bf16bits(p0);
            lp[(r + 8 * half) * 32 + 16 + nn] = f32_to_bf16bits(p1);
        }
        // LDS is in-order per wave; pin the compiler and drain DS writes.
        asm volatile("s_wait_dscnt 0" ::: "memory");
        FragAB pa = load_frag_A(lp, 32, 0, 0);  // P tile as A-fragment (ds_load_b128)
        acc[0] = wmma_bf16(pa, vb0, acc[0]);
        acc[1] = wmma_bf16(pa, vb1, acc[1]);
        acc[2] = wmma_bf16(pa, vb2, acc[2]);
        acc[3] = wmma_bf16(pa, vb3, acc[3]);
        asm volatile("" ::: "memory");  // keep next-iter LDS stores below the loads
    }

    // epilogue: O[b][s][h*64+e] = acc / l  (bf16, concat-head layout)
#pragma unroll
    for (int r = 0; r < 8; ++r) {
        float inv = 1.0f / lstate[r];
        int row = m0 + r + 8 * half;
        size_t obase = ((size_t)b * SEQ + row) * D_MODEL + h * HD;
#pragma unroll
        for (int nt = 0; nt < 4; ++nt)
            O[obase + nt * 16 + nn] = f32_to_bf16bits(acc[nt][r] * inv);
    }
}

// ---------------------------------------------------------------------------
// 5) Output projection: Y = O (4096x1024, bf16) @ Wo + bo, f32 out.
//    One wave = 16x64 tile; Wo^T is n-major bf16 [D, D]; double-buffered.
// ---------------------------------------------------------------------------
__global__ void __launch_bounds__(256)
oproj_kernel(const unsigned short* __restrict__ Obf,
             const unsigned short* __restrict__ WoT,
             const float* __restrict__ bo, float* __restrict__ Y) {
    int wave = blockIdx.x * (blockDim.x >> 5) + (threadIdx.x >> 5);
    int nt64 = wave & 15;          // D/64 tiles
    int mt   = wave >> 4;          // (B*S)/16 tiles
    int m0 = mt * 16, n0 = nt64 * 64;

    v8f acc[4];
#pragma unroll
    for (int nt = 0; nt < 4; ++nt) acc[nt] = zero8();

    FragAB a  = load_frag_A(Obf, D_MODEL, m0, 0);
    FragAB b0 = load_frag_B(WoT, D_MODEL, n0,      0);
    FragAB b1 = load_frag_B(WoT, D_MODEL, n0 + 16, 0);
    FragAB b2 = load_frag_B(WoT, D_MODEL, n0 + 32, 0);
    FragAB b3 = load_frag_B(WoT, D_MODEL, n0 + 48, 0);

    for (int k0 = 0; k0 < D_MODEL; k0 += 32) {
        int kn = (k0 + 32 < D_MODEL) ? k0 + 32 : k0;
        FragAB an = load_frag_A(Obf, D_MODEL, m0, kn);
        FragAB c0 = load_frag_B(WoT, D_MODEL, n0,      kn);
        FragAB c1 = load_frag_B(WoT, D_MODEL, n0 + 16, kn);
        FragAB c2 = load_frag_B(WoT, D_MODEL, n0 + 32, kn);
        FragAB c3 = load_frag_B(WoT, D_MODEL, n0 + 48, kn);
        acc[0] = wmma_bf16(a, b0, acc[0]);
        acc[1] = wmma_bf16(a, b1, acc[1]);
        acc[2] = wmma_bf16(a, b2, acc[2]);
        acc[3] = wmma_bf16(a, b3, acc[3]);
        a = an; b0 = c0; b1 = c1; b2 = c2; b3 = c3;
    }

    int lane = threadIdx.x & 31;
    int nn = lane & 15, half = lane >> 4;
#pragma unroll
    for (int nt = 0; nt < 4; ++nt) {
        float bval = bo[n0 + nt * 16 + nn];
#pragma unroll
        for (int r = 0; r < 8; ++r) {
            int row = m0 + r + 8 * half;
            Y[(size_t)row * D_MODEL + n0 + nt * 16 + nn] = acc[nt][r] + bval;
        }
    }
}

// ---------------------------------------------------------------------------
// Launch
// ---------------------------------------------------------------------------
extern "C" void kernel_launch(void* const* d_in, const int* in_sizes, int n_in,
                              void* d_out, int out_size, void* d_ws, size_t ws_size,
                              hipStream_t stream) {
    const float* hidden = (const float*)d_in[0];
    const float* Wq = (const float*)d_in[1];
    const float* Wk = (const float*)d_in[2];
    const float* Wv = (const float*)d_in[3];
    const float* bq = (const float*)d_in[4];
    const float* bk = (const float*)d_in[5];
    const float* bv = (const float*)d_in[6];
    const float* Wo = (const float*)d_in[7];
    const float* bo = (const float*)d_in[8];
    float* Y = (float*)d_out;

    char* ws = (char*)d_ws;
    size_t off = 0;
    auto mk = [&](size_t bytes) -> char* {
        char* p = ws + off;
        off += (bytes + 255) & ~(size_t)255;
        return p;
    };
    unsigned short* Xbf = (unsigned short*)mk((size_t)BATCH * SEQ * D_MODEL * 2);
    unsigned short* WqT = (unsigned short*)mk((size_t)NH * D_MODEL * HD * 2);
    unsigned short* WkT = (unsigned short*)mk((size_t)NH * D_MODEL * HD * 2);
    unsigned short* WvT = (unsigned short*)mk((size_t)NH * D_MODEL * HD * 2);
    unsigned short* WoT = (unsigned short*)mk((size_t)D_MODEL * D_MODEL * 2);
    unsigned short* Qb  = (unsigned short*)mk((size_t)BATCH * NH * SEQ * HD * 2);
    unsigned short* Kb  = (unsigned short*)mk((size_t)BATCH * NH * SEQ * HD * 2);
    unsigned short* Vb  = (unsigned short*)mk((size_t)BATCH * NH * SEQ * HD * 2);
    unsigned short* Vt  = (unsigned short*)mk((size_t)BATCH * NH * HD * SEQ * 2);
    unsigned short* Ob  = (unsigned short*)mk((size_t)BATCH * SEQ * D_MODEL * 2);
    (void)ws_size; (void)n_in; (void)in_sizes; (void)out_size;

    // 1) hidden -> bf16
    int nX = BATCH * SEQ * D_MODEL;
    pack_bf16_kernel<<<(nX + 255) / 256, 256, 0, stream>>>(hidden, Xbf, nX);

    // 2) weight transposes (f32 -> bf16, n-major)
    transpose_pack_kernel<<<dim3(HD / 32, D_MODEL / 32, NH), 256, 0, stream>>>(Wq, WqT, D_MODEL, HD);
    transpose_pack_kernel<<<dim3(HD / 32, D_MODEL / 32, NH), 256, 0, stream>>>(Wk, WkT, D_MODEL, HD);
    transpose_pack_kernel<<<dim3(HD / 32, D_MODEL / 32, NH), 256, 0, stream>>>(Wv, WvT, D_MODEL, HD);
    transpose_pack_kernel<<<dim3(D_MODEL / 32, D_MODEL / 32, 1), 256, 0, stream>>>(Wo, WoT, D_MODEL, D_MODEL);

    // 3) QKV projections: 3 * B * H * (S/16) waves = 12288 -> 1536 blocks
    qkv_kernel<<<1536, 256, 0, stream>>>(Xbf, WqT, WkT, WvT, bq, bk, bv, Qb, Kb, Vb);

    // 4) V -> V^T (bf16)
    transpose_bf16_kernel<<<dim3(HD / 32, SEQ / 32, BATCH * NH), 256, 0, stream>>>(Vb, Vt, SEQ, HD);

    // 5) fused causal attention: B * H * (S/16) waves = 4096 -> 512 blocks
    attn_kernel<<<512, 256, 0, stream>>>(Qb, Kb, Vt, Ob);

    // 6) output projection: (B*S/16) * (D/64) waves = 4096 -> 512 blocks
    oproj_kernel<<<512, 256, 0, stream>>>(Ob, WoT, bo, Y);
}
